// MultiStep_16707422781440
// MI455X (gfx1250) — compile-verified, matrix operand order checked by reading, not run on hardware
//
#include <hip/hip_runtime.h>
#include <hip/hip_bf16.h>
#include <stdint.h>

// Problem constants (match reference)
#define GAMMA_F 0.99f
#define NSTEP 3
#define BB 2048
#define TT 512
#define DD 64

// Output layout: tuple concatenated flat in return order, all as float32.
#define OFF_GM ((size_t)0)                                   // gamma_masked      (B,T,1)
#define OFF_ST ((size_t)BB * TT)                             // steps_to_next_obs (B,T,1)
#define OFF_NT ((size_t)2 * BB * TT)                         // nonterminal       (B,T,1)
#define OFF_PR ((size_t)3 * BB * TT)                         // partial_return    (B,T,1)
#define OFF_NO ((size_t)4 * BB * TT)                         // next_obs_out      (B,T,D)
#define OFF_DO ((size_t)4 * BB * TT + (size_t)BB * TT * DD)  // done_out          (B,T,1)

typedef __attribute__((ext_vector_type(16))) _Float16 v16h;
typedef __attribute__((ext_vector_type(8)))  float    v8f;

__global__ __launch_bounds__(256) void multistep_kernel(
    const unsigned char* __restrict__ done,
    const unsigned char* __restrict__ mask,
    const float* __restrict__ reward,
    const float* __restrict__ obs,
    float* __restrict__ out)
{
  const int b   = blockIdx.x;
  const int tid = threadIdx.x;
  const size_t rb = (size_t)b * TT;

  __shared__ float s_rew[TT + 16];        // reward padded with zeros
  __shared__ float s_done[TT];            // done as 0/1
  __shared__ float s_mask[TT];            // mask as 0/1
  __shared__ float s_pt[TT + NSTEP];      // post_terminal (extended with 1s)
  __shared__ float s_steps[TT];           // clipped suffix counts
  __shared__ float s_last[DD];            // obs at terminal (sum over terminals)
  __shared__ int   s_termList[32];
  __shared__ int   s_termCount;
  __shared__ float s_doneLast;

  // ---------------- Phase 0: stage per-row inputs into LDS ----------------
  // reward via CDNA5 async global->LDS DMA (ASYNCcnt path), 2 x b32 per thread
  {
    unsigned lds0 = (unsigned)(uintptr_t)(&s_rew[tid]);
    unsigned lds1 = (unsigned)(uintptr_t)(&s_rew[tid + 256]);
    unsigned long long g0 = (unsigned long long)(uintptr_t)(reward + rb + tid);
    unsigned long long g1 = (unsigned long long)(uintptr_t)(reward + rb + tid + 256);
    asm volatile("global_load_async_to_lds_b32 %0, %1, off" :: "v"(lds0), "v"(g0) : "memory");
    asm volatile("global_load_async_to_lds_b32 %0, %1, off" :: "v"(lds1), "v"(g1) : "memory");
  }
  // done / mask bytes -> 0/1 floats
  for (int t = tid; t < TT; t += 256) {
    s_done[t] = done[rb + t] ? 1.0f : 0.0f;
    s_mask[t] = mask[rb + t] ? 1.0f : 0.0f;
  }
  if (tid < 16)    s_rew[TT + tid] = 0.0f;  // zero pad r[T..T+15]
  if (tid < NSTEP) s_pt[TT + tid]  = 1.0f;  // post_terminal extension
  asm volatile("s_wait_asynccnt 0x0" ::: "memory");
  __syncthreads();

  // ---------------- Phase 1: per-row scans (thread 0, in LDS) ----------------
  if (tid == 0) {
    float S = 0.f;
    for (int t = 0; t < TT; ++t) S += s_done[t];
    const int cnt = (int)(S + 0.5f);

    int nterm = 0, c1 = 0, c2 = 0;
    float dlast = 0.f;
    for (int t = 0; t < TT; ++t) {
      int termi;
      if (t == TT - 1) termi = (s_done[t] > 0.5f || cnt != 1) ? 1 : 0;
      else             termi = (s_done[t] > 0.5f) ? 1 : 0;
      if (termi) {
        if (nterm < 32) s_termList[nterm] = t;
        ++nterm;
        dlast += s_done[t];                // done_last = sum(terminal * done)
      }
      c1 += termi;
      c2 += c1;
      s_pt[t] = (c2 >= 2) ? 1.0f : 0.0f;   // pt = double-cumsum >= 2
    }
    s_termCount = (nterm < 32) ? nterm : 32;
    s_doneLast  = dlast;

    int st = 0;                             // suffix-sum of nonterminal, clipped
    for (int t = TT - 1; t >= 0; --t) {
      if (s_pt[t] < 0.5f) ++st;
      s_steps[t] = (float)((st < NSTEP + 1) ? st : (NSTEP + 1));
    }
  }
  __syncthreads();

  // ---------------- Phase 2: obs at terminal position(s) ----------------
  if (tid < DD) {
    float acc = 0.f;
    const int n = s_termCount;
    for (int i = 0; i < n; ++i)
      acc += obs[(rb + (size_t)s_termList[i]) * DD + tid];
    s_last[tid] = acc;
  }

  // ---------------- Phase 3a: small per-t outputs ----------------
  const float dlast = s_doneLast;
  for (int t = tid; t < TT; t += 256) {
    float gm = 1.0f;
#pragma unroll
    for (int k = 0; k <= NSTEP; ++k) {
      const int u = t + k;
      gm *= (u < TT && s_mask[u] > 0.5f) ? GAMMA_F : 1.0f;
    }
    out[OFF_GM + rb + t] = gm;
    out[OFF_ST + rb + t] = s_steps[t];
    out[OFF_NT + rb + t] = (s_pt[t] < 0.5f) ? 1.0f : 0.0f;
    const float cat = ((t + NSTEP < TT) ? s_done[t + NSTEP] : 0.0f)
                    + dlast * s_pt[t + NSTEP];
    out[OFF_DO + rb + t] = (s_mask[t] > 0.5f && cat > 0.5f) ? 1.0f : 0.0f;
  }

  // ---------------- Phase 3b: partial_return via WMMA (wave 0) ----------------
  // D = A(16x32 f16) x B(32x16 f16) + 0 : A[m,k]=r[base+m+k] (k<4), B[k,n]=gamma^k (k<4)
  // All columns of D are equal; D[m,n] = partial_return[base+m].
  if (tid < 32) {
    const int lane = tid;
    v16h bmat = {};
    if (lane < 16) {
      float g = 1.0f;
#pragma unroll
      for (int k = 0; k <= NSTEP; ++k) { bmat[k] = (_Float16)g; g *= GAMMA_F; }
    }
    for (int base = 0; base < TT; base += 16) {
      v16h amat = {};
      if (lane < 16) {
#pragma unroll
        for (int k = 0; k <= NSTEP; ++k)
          amat[k] = (_Float16)s_rew[base + lane + k];
      }
      v8f acc = {};
      acc = __builtin_amdgcn_wmma_f32_16x16x32_f16(
          /*neg_a=*/false, amat, /*neg_b=*/false, bmat,
          /*c_mod=*/(short)0, acc, /*reuse_a=*/false, /*reuse_b=*/false);
      // C/D layout: VGPR i, lanes 0-15 -> M=i ; lanes 16-31 -> M=8+i
      if (lane == 0) {
#pragma unroll
        for (int i = 0; i < 8; ++i) out[OFF_PR + rb + base + i] = acc[i];
      } else if (lane == 16) {
#pragma unroll
        for (int i = 0; i < 8; ++i) out[OFF_PR + rb + base + 8 + i] = acc[i];
      }
    }
  }
  __syncthreads();

  // ---------------- Phase 4: stream next_obs_out (memory-bound main loop) ----------------
  // out[t,d] = mask[t] ? (t+3<T ? obs[t+3,d] : 0) + (pt_ext[t+3] ? last[d] : 0) : 0
  const int dq = (tid & 15) * 4;            // float4 chunk within the row
  for (int t = tid >> 4; t < TT; t += 16) {
    float4 v = make_float4(0.f, 0.f, 0.f, 0.f);
    if (s_mask[t] > 0.5f) {
      if (t + NSTEP < TT) {
        const float* p = obs + (rb + t + NSTEP) * DD + dq;
        __builtin_prefetch(p + 16 * DD, 0, 0);   // global_prefetch_b8
        v = *(const float4*)p;
      }
      if (s_pt[t + NSTEP] > 0.5f) {
        v.x += s_last[dq + 0];
        v.y += s_last[dq + 1];
        v.z += s_last[dq + 2];
        v.w += s_last[dq + 3];
      }
    }
    *(float4*)(out + OFF_NO + (rb + t) * DD + dq) = v;
  }
}

extern "C" void kernel_launch(void* const* d_in, const int* in_sizes, int n_in,
                              void* d_out, int out_size, void* d_ws, size_t ws_size,
                              hipStream_t stream) {
  (void)in_sizes; (void)n_in; (void)out_size; (void)d_ws; (void)ws_size;
  const unsigned char* done   = (const unsigned char*)d_in[0]; // bool (1B)
  const unsigned char* mask   = (const unsigned char*)d_in[1]; // bool (1B)
  const float*         reward = (const float*)d_in[2];
  const float*         obs    = (const float*)d_in[3];
  float*               out    = (float*)d_out;
  multistep_kernel<<<dim3(BB), dim3(256), 0, stream>>>(done, mask, reward, obs, out);
}
